// _PureMamba_54528904790904
// MI455X (gfx1250) — matrix operand order repaired and checked by reading, not compiled, not run on hardware
//
#include <hip/hip_runtime.h>

// ---------------- problem sizes ----------------
#define B_SZ   2
#define SEQ_L  2048
#define DMODEL 1024
#define DINNER 2048
#define DSTATE 16
#define DCONV  4
#define BL     (B_SZ * SEQ_L)          // 4096 rows
#define EPROJ  (2 * DINNER)            // 4096 (xz width)

// ---------------- types for WMMA ----------------
typedef __attribute__((ext_vector_type(16))) __bf16     v16bf;
typedef __attribute__((ext_vector_type(8)))  float      v8f;
typedef __attribute__((ext_vector_type(4)))  unsigned int u32x4;

union Frag { u32x4 u[2]; v16bf bf; };

__device__ __forceinline__ unsigned short f2bf(float f) {
    unsigned int u = __float_as_uint(f);
    u += 0x7FFFu + ((u >> 16) & 1u);          // round-to-nearest-even
    return (unsigned short)(u >> 16);
}

// ---------------- fp32 -> bf16 convert ----------------
__global__ __launch_bounds__(256) void k_f32_to_bf16(const float* __restrict__ in,
                                                     unsigned short* __restrict__ out,
                                                     int n) {
    int i = blockIdx.x * 256 + threadIdx.x;
    if (i < n) out[i] = f2bf(in[i]);
}

// ---------------- bf16 WMMA GEMM with async-to-LDS double buffering ----------------
//  C[M,N] = A[M,K] * Bw[N,K]^T
//  WG tile 128x128 (8 waves, 2x4), wave tile 64x32 (4x2 WMMA tiles), K step 32.
//  Tiles staged in LDS via GLOBAL_LOAD_ASYNC_TO_LDS_B128 (ASYNCcnt), rows padded
//  to 80B so fragment ds_load_b128 is bank-conflict-free.
#define LDS_ROW 40   // 32 data halfs + 8 pad halfs = 80 bytes
__global__ __launch_bounds__(256) void k_gemm_bf16_lds(const unsigned short* __restrict__ A,
                                                       const unsigned short* __restrict__ Bw,
                                                       float* __restrict__ C,
                                                       int M, int N, int K) {
    __shared__ __align__(16) unsigned short sA[2][128 * LDS_ROW];
    __shared__ __align__(16) unsigned short sB[2][128 * LDS_ROW];

    const int tid  = threadIdx.x;
    const int wave = tid >> 5;
    const int lane = tid & 31;
    const int r    = lane & 15;
    const int half = lane >> 4;
    const int wm   = wave >> 2;          // 0..1
    const int wn   = wave & 3;           // 0..3
    const int m0   = blockIdx.x * 128;
    const int n0   = blockIdx.y * 128;

    // fill mapping: 512 16B chunks per tile, 256 threads -> 2 chunks each (rows c, c+64)
    const int crow = tid >> 2;           // 0..63
    const int ccol = tid & 3;            // 0..3 (16B chunk within 64B k-row)
    const unsigned short* gA0 = A  + (size_t)(m0 + crow)      * K + ccol * 8;
    const unsigned short* gA1 = A  + (size_t)(m0 + crow + 64) * K + ccol * 8;
    const unsigned short* gB0 = Bw + (size_t)(n0 + crow)      * K + ccol * 8;
    const unsigned short* gB1 = Bw + (size_t)(n0 + crow + 64) * K + ccol * 8;

    auto fill = [&](int buf, int k0) {
        unsigned la0 = (unsigned)(size_t)&sA[buf][crow * LDS_ROW + ccol * 8];
        unsigned la1 = la0 + 64 * LDS_ROW * 2;
        unsigned lb0 = (unsigned)(size_t)&sB[buf][crow * LDS_ROW + ccol * 8];
        unsigned lb1 = lb0 + 64 * LDS_ROW * 2;
        unsigned long long a0 = (unsigned long long)(gA0 + k0);
        unsigned long long a1 = (unsigned long long)(gA1 + k0);
        unsigned long long b0 = (unsigned long long)(gB0 + k0);
        unsigned long long b1 = (unsigned long long)(gB1 + k0);
        asm volatile("global_load_async_to_lds_b128 %0, %1, off" :: "v"(la0), "v"(a0) : "memory");
        asm volatile("global_load_async_to_lds_b128 %0, %1, off" :: "v"(la1), "v"(a1) : "memory");
        asm volatile("global_load_async_to_lds_b128 %0, %1, off" :: "v"(lb0), "v"(b0) : "memory");
        asm volatile("global_load_async_to_lds_b128 %0, %1, off" :: "v"(lb1), "v"(b1) : "memory");
    };

    v8f acc[4][2];
    v8f zero = {};
#pragma unroll
    for (int i = 0; i < 4; ++i)
#pragma unroll
        for (int j = 0; j < 2; ++j) acc[i][j] = zero;

    const int KS = K >> 5;
    fill(0, 0);
    asm volatile("s_wait_asynccnt 0" ::: "memory");
    __syncthreads();

    for (int kt = 0; kt < KS; ++kt) {
        const int cur = kt & 1;
        if (kt + 1 < KS) fill(1 - cur, (kt + 1) << 5);   // overlap next-stage DMA with compute

        const unsigned short* As = &sA[cur][0];
        const unsigned short* Bs = &sB[cur][0];
        // A fragments: lane r holds K {0..7,16..23}, lane r+16 holds K {8..15,24..31}
        Frag a[4];
#pragma unroll
        for (int i = 0; i < 4; ++i) {
            const unsigned short* p = As + (wm * 64 + 16 * i + r) * LDS_ROW + half * 8;
            a[i].u[0] = *(const u32x4*)p;
            a[i].u[1] = *(const u32x4*)(p + 16);
        }
        // B fragments: lane r holds K {0..7,16..23}, lane r+16 holds K {8..15,24..31}
        Frag b[2];
#pragma unroll
        for (int j = 0; j < 2; ++j) {
            const unsigned short* p = Bs + (wn * 32 + 16 * j + r) * LDS_ROW + half * 8;
            b[j].u[0] = *(const u32x4*)p;
            b[j].u[1] = *(const u32x4*)(p + 16);
        }
#pragma unroll
        for (int i = 0; i < 4; ++i)
#pragma unroll
            for (int j = 0; j < 2; ++j)
                acc[i][j] = __builtin_amdgcn_wmma_f32_16x16x32_bf16(
                    false, a[i].bf, false, b[j].bf, (short)0, acc[i][j], false, false);

        if (kt + 1 < KS) { asm volatile("s_wait_asynccnt 0" ::: "memory"); }
        __syncthreads();
    }

    // D layout: VGPR rr -> M = rr (lanes 0-15) / 8+rr (lanes 16-31); N = lane&15
#pragma unroll
    for (int i = 0; i < 4; ++i)
#pragma unroll
        for (int j = 0; j < 2; ++j)
#pragma unroll
            for (int rr = 0; rr < 8; ++rr)
                C[(size_t)(m0 + wm * 64 + 16 * i + half * 8 + rr) * N + n0 + wn * 32 + 16 * j + r] =
                    acc[i][j][rr];
}

// ---------------- causal depthwise conv (DCONV=4) + bias + SiLU ----------------
__global__ __launch_bounds__(256) void k_conv_silu(const float* __restrict__ xz,
                                                   const float* __restrict__ conv_w,
                                                   const float* __restrict__ conv_b,
                                                   float* __restrict__ xc) {
    int idx = blockIdx.x * 256 + threadIdx.x;      // over BL*DINNER
    if (idx >= BL * DINNER) return;
    int d = idx & (DINNER - 1);
    int m = idx >> 11;                             // b*L + l
    int l = m & (SEQ_L - 1);
    float acc = conv_b[d];
#pragma unroll
    for (int k = 0; k < DCONV; ++k) {
        int ll = l + k - (DCONV - 1);
        if (ll >= 0) acc += conv_w[d * DCONV + k] * xz[(size_t)(m + k - (DCONV - 1)) * EPROJ + d];
    }
    xc[(size_t)m * DINNER + d] = acc / (1.f + __expf(-acc));   // SiLU
}

// ---------------- x_proj: bcd[m, 0..32] = xc[m,:] . x_proj_w[j,:] ----------------
__global__ __launch_bounds__(256) void k_xproj(const float* __restrict__ xc,
                                               const float* __restrict__ xpw,
                                               float* __restrict__ bcd) {
    __shared__ float row[DINNER];
    int m = blockIdx.x;
    const float* src = xc + (size_t)m * DINNER;
    for (int i = threadIdx.x; i < DINNER; i += 256) row[i] = src[i];
    __syncthreads();
    int wave = threadIdx.x >> 5, lane = threadIdx.x & 31;
    for (int j = wave; j < 2 * DSTATE + 1; j += 8) {
        const float* w = xpw + (size_t)j * DINNER;
        float s = 0.f;
        for (int i = lane; i < DINNER; i += 32) s = fmaf(row[i], w[i], s);
#pragma unroll
        for (int o = 16; o > 0; o >>= 1) s += __shfl_xor(s, o, 32);
        if (lane == 0) bcd[(size_t)m * 33 + j] = s;
    }
}

// ---------------- selective scan, fused softplus / +x*D / silu(z) gate / bf16 out ----
__global__ __launch_bounds__(256) void k_scan(const float* __restrict__ bcd,
                                              const float* __restrict__ xc,
                                              const float* __restrict__ xz,
                                              const float* __restrict__ A_log,
                                              const float* __restrict__ dt_w,
                                              const float* __restrict__ dt_b,
                                              const float* __restrict__ Dp,
                                              unsigned short* __restrict__ ygbf) {
    int n    = threadIdx.x & 15;
    int chan = blockIdx.x * (256 / 16) + (threadIdx.x >> 4);   // [0, B*DINNER)
    int b = chan >> 11;
    int d = chan & (DINNER - 1);

    float An = -__expf(A_log[d * DSTATE + n]);
    float w  = dt_w[d], bb = dt_b[d], Dd = Dp[d];
    float h  = 0.f;

    const float* bc   = bcd + (size_t)b * SEQ_L * 33;
    const float* xrow = xc  + (size_t)b * SEQ_L * DINNER + d;
    const float* zrow = xz  + (size_t)b * SEQ_L * EPROJ + DINNER + d;
    unsigned short* yrow = ygbf + (size_t)b * SEQ_L * DINNER + d;

    for (int t = 0; t < SEQ_L; ++t) {
        if (t + 1 < SEQ_L) {                       // global_prefetch_b8 on gfx1250
            __builtin_prefetch(xrow + (size_t)(t + 1) * DINNER, 0, 1);
            __builtin_prefetch(zrow + (size_t)(t + 1) * EPROJ, 0, 1);
        }
        float delta = bc[t * 33 + 32];
        float Bn    = bc[t * 33 + n];
        float Cn    = bc[t * 33 + DSTATE + n];
        float xv    = xrow[(size_t)t * DINNER];

        float u  = fmaf(delta, w, bb);
        float dt = (u > 20.f) ? u : log1pf(__expf(u));     // softplus
        float dA = __expf(dt * An);
        h = fmaf(dA, h, dt * xv * Bn);
        float p = h * Cn;
        p += __shfl_xor(p, 1, 32);
        p += __shfl_xor(p, 2, 32);
        p += __shfl_xor(p, 4, 32);
        p += __shfl_xor(p, 8, 32);                 // sum over 16 states
        if (n == 0) {
            float z  = zrow[(size_t)t * EPROJ];
            float sz = z / (1.f + __expf(-z));     // silu(z)
            float y  = (p + xv * Dd) * sz;
            yrow[(size_t)t * DINNER] = f2bf(y);
        }
    }
}

// ---------------- launcher ----------------
extern "C" void kernel_launch(void* const* d_in, const int* in_sizes, int n_in,
                              void* d_out, int out_size, void* d_ws, size_t ws_size,
                              hipStream_t stream) {
    const float* x        = (const float*)d_in[0];
    const float* in_w     = (const float*)d_in[1];
    const float* conv_w   = (const float*)d_in[2];
    const float* conv_b   = (const float*)d_in[3];
    const float* xproj_w  = (const float*)d_in[4];
    const float* dt_w     = (const float*)d_in[5];
    const float* dt_b     = (const float*)d_in[6];
    const float* out_w    = (const float*)d_in[7];
    const float* A_log    = (const float*)d_in[8];
    const float* D_param  = (const float*)d_in[9];
    float* out = (float*)d_out;

    // workspace carve-up
    char* ws = (char*)d_ws;
    float*          xz    = (float*)ws;                       ws += (size_t)BL * EPROJ * 4;      // 67 MB
    float*          xc    = (float*)ws;                       ws += (size_t)BL * DINNER * 4;     // 33.5 MB
    float*          bcd   = (float*)ws;                       ws += (size_t)BL * 33 * 4;         // 0.54 MB
    unsigned short* ygbf  = (unsigned short*)ws;              ws += (size_t)BL * DINNER * 2;     // 16.8 MB
    unsigned short* xbf   = (unsigned short*)ws;              ws += (size_t)BL * DMODEL * 2;     // 8.4 MB
    unsigned short* winbf = (unsigned short*)ws;              ws += (size_t)EPROJ * DMODEL * 2;  // 8.4 MB
    unsigned short* woutbf= (unsigned short*)ws;              ws += (size_t)DMODEL * DINNER * 2; // 4.2 MB

    // 1) bf16 conversions
    {
        int n1 = BL * DMODEL;
        k_f32_to_bf16<<<(n1 + 255) / 256, 256, 0, stream>>>(x, xbf, n1);
        int n2 = EPROJ * DMODEL;
        k_f32_to_bf16<<<(n2 + 255) / 256, 256, 0, stream>>>(in_w, winbf, n2);
        int n3 = DMODEL * DINNER;
        k_f32_to_bf16<<<(n3 + 255) / 256, 256, 0, stream>>>(out_w, woutbf, n3);
    }
    // 2) in_proj GEMM: xz[BL, 4096] = xbf[BL,1024] * winbf[4096,1024]^T
    k_gemm_bf16_lds<<<dim3(BL / 128, EPROJ / 128), 256, 0, stream>>>(xbf, winbf, xz, BL, EPROJ, DMODEL);
    // 3) depthwise causal conv + SiLU
    k_conv_silu<<<(BL * DINNER + 255) / 256, 256, 0, stream>>>(xz, conv_w, conv_b, xc);
    // 4) x_proj -> bcd
    k_xproj<<<BL, 256, 0, stream>>>(xc, xproj_w, bcd);
    // 5) selective scan (fused dt/softplus, +x*D, silu(z) gate, bf16 cast)
    k_scan<<<(B_SZ * DINNER) / 16, 256, 0, stream>>>(bcd, xc, xz, A_log, dt_w, dt_b, D_param, ygbf);
    // 6) out_proj GEMM: out[BL,1024] = ygbf[BL,2048] * woutbf[1024,2048]^T
    k_gemm_bf16_lds<<<dim3(BL / 128, DMODEL / 128), 256, 0, stream>>>(ygbf, woutbf, out, BL, DMODEL, DINNER);
}
